// Vote_Encoder_2370821948159
// MI455X (gfx1250) — matrix operand order repaired and checked by reading, not compiled, not run on hardware
//
#include <hip/hip_runtime.h>
#include <cstdint>
#include <cstddef>

// ---------------------------------------------------------------------------
// MI455X / gfx1250 implementation. wave32, WMMA 16x16x32 bf16 (f32 accum).
// All GEMM operands are bf16; weights pre-packed transposed (N x K) so each
// B fragment is two contiguous 16B loads matching the ISA B-matrix VGPR
// layout. A fragments load directly from row-major bf16 activations.
// ---------------------------------------------------------------------------

using bf16 = __bf16;
typedef bf16  bf16x16 __attribute__((ext_vector_type(16)));
typedef float f32x8   __attribute__((ext_vector_type(8)));
typedef unsigned int u32x4 __attribute__((ext_vector_type(4)));

#define LEAKY_SLOPE 0.1f

// ---------------- packing / conversion ----------------
__global__ void pack_wt_kernel(const float* __restrict__ W, bf16* __restrict__ Wt,
                               int K, int N) {
  int i = blockIdx.x * blockDim.x + threadIdx.x;
  if (i >= K * N) return;
  int k = i / N, n = i - k * N;
  Wt[(size_t)n * K + k] = (bf16)W[i];          // transpose + bf16
}

__global__ void f32_to_bf16_kernel(const float* __restrict__ X, bf16* __restrict__ Y,
                                   size_t n) {
  size_t i = blockIdx.x * (size_t)blockDim.x + threadIdx.x;
  if (i < n) Y[i] = (bf16)X[i];
}

// ---------------- generic WMMA GEMM ----------------
// C[M,N] = A[M,K](bf16,row-major) * W[K,N]  with W pre-packed as Wt[N,K] bf16.
// Block = 128 threads = 4 waves; each wave owns a 32x64 register tile
// (2 A fragments x 4 B fragments = 8 WMMAs per k-step, B loads amortized).
// act: 0 none, 1 relu, 2 leaky.  Cf (f32) and/or Cb (bf16) outputs.
__global__ void __launch_bounds__(128)
gemm_bf16_wmma(const bf16* __restrict__ A, const bf16* __restrict__ Wt,
               const float* __restrict__ bias,
               float* __restrict__ Cf, bf16* __restrict__ Cb,
               int N, int K, int act)
{
  const int lane = threadIdx.x & 31;
  const int wave = threadIdx.x >> 5;
  const int l16  = lane & 15;
  const int half = lane >> 4;                 // 0: K 0..7 / 16..23 ; 1: +8
  const int rowBase = (blockIdx.x * 4 + wave) * 32;
  const int colBase = blockIdx.y * 64;

  const bf16* Ap0 = A + (size_t)(rowBase + l16) * K + half * 8;
  const bf16* Ap1 = Ap0 + (size_t)16 * K;
  const bf16* Wp[4];
#pragma unroll
  for (int j = 0; j < 4; ++j)
    Wp[j] = Wt + (size_t)(colBase + j * 16 + l16) * K + half * 8;

  f32x8 acc0[4] = {};
  f32x8 acc1[4] = {};

  for (int k0 = 0; k0 < K; k0 += 32) {
    // hoist all fragment loads so waits can be staggered against the WMMAs
    union { u32x4 u[2]; bf16x16 v; } a0, a1, b[4];
    a0.u[0] = *(const u32x4*)(Ap0 + k0);
    a0.u[1] = *(const u32x4*)(Ap0 + k0 + 16);
    a1.u[0] = *(const u32x4*)(Ap1 + k0);
    a1.u[1] = *(const u32x4*)(Ap1 + k0 + 16);
#pragma unroll
    for (int j = 0; j < 4; ++j) {
      b[j].u[0] = *(const u32x4*)(Wp[j] + k0);
      b[j].u[1] = *(const u32x4*)(Wp[j] + k0 + 16);
    }
    __builtin_prefetch(Ap0 + k0 + 512, 0, 1);  // streamed A into nearer cache
    __builtin_prefetch(Ap1 + k0 + 512, 0, 1);
#pragma unroll
    for (int j = 0; j < 4; ++j) {
      acc0[j] = __builtin_amdgcn_wmma_f32_16x16x32_bf16(false, a0.v, false, b[j].v,
                                                        (short)0, acc0[j], false, false);
      acc1[j] = __builtin_amdgcn_wmma_f32_16x16x32_bf16(false, a1.v, false, b[j].v,
                                                        (short)0, acc1[j], false, false);
    }
  }

#pragma unroll
  for (int j = 0; j < 4; ++j) {
    int col = colBase + j * 16 + l16;
    float bv = bias ? bias[col] : 0.0f;
#pragma unroll
    for (int r = 0; r < 8; ++r) {
      int row = rowBase + half * 8 + r;       // C/D VGPR layout
      float v0 = acc0[j][r] + bv;
      float v1 = acc1[j][r] + bv;
      if (act == 1)      { v0 = fmaxf(v0, 0.0f); v1 = fmaxf(v1, 0.0f); }
      else if (act == 2) { v0 = (v0 >= 0.0f) ? v0 : LEAKY_SLOPE * v0;
                           v1 = (v1 >= 0.0f) ? v1 : LEAKY_SLOPE * v1; }
      size_t o0 = (size_t)row * N + col;
      size_t o1 = (size_t)(row + 16) * N + col;
      if (Cf) { Cf[o0] = v0; Cf[o1] = v1; }
      if (Cb) { Cb[o0] = (bf16)v0; Cb[o1] = (bf16)v1; }
    }
  }
}

// ---------------- KPConv aggregation (stage 1) ----------------
// One wave per query point m. Computes agg[m,k,c] = sum_h infl[m,h,k]*f[nbr[m,h],c]
// as a 16x64 (padded) x 64xCmid WMMA product. Neighbor features are staged
// *transposed* in LDS so B fragments are contiguous. Output written bf16 in
// (k*Cmid + c) order, ready to be the A matrix of the stage-2 GEMM.
__global__ void __launch_bounds__(32)
kpconv_agg_kernel(const float* __restrict__ S, int Ns,      // support pts, pad id==Ns -> 1e6
                  const float* __restrict__ Q,              // query centers Mx3
                  const bf16* __restrict__ F, int Cmid,     // features Ns x Cmid, pad -> 0
                  const int* __restrict__ nbr,              // M x 38
                  const float* __restrict__ kpts, float sigma,
                  bf16* __restrict__ Agg)                   // M x (15*Cmid)
{
  __shared__ int   idx[64];
  __shared__ float relx[64], rely[64], relz[64];
  __shared__ __align__(32) bf16 Bt[128 * 64];               // [channel][neighbor]

  const int m    = blockIdx.x;
  const int lane = threadIdx.x;
  const float qx = Q[(size_t)m*3+0], qy = Q[(size_t)m*3+1], qz = Q[(size_t)m*3+2];

  for (int h = lane; h < 64; h += 32) {
    int id = (h < 38) ? nbr[(size_t)m*38 + h] : Ns;
    idx[h] = id;
    float sx, sy, sz;
    if (id < Ns) { sx = S[(size_t)id*3]; sy = S[(size_t)id*3+1]; sz = S[(size_t)id*3+2]; }
    else         { sx = sy = sz = 1.0e6f; }
    relx[h] = sx - qx; rely[h] = sy - qy; relz[h] = sz - qz;
  }
  __syncthreads();

  // gather neighbor features (bf16) transposed into LDS; zero-pad h>=38 / pad row
  for (int h = lane; h < 64; h += 32) {
    int id = idx[h];
    bool valid = (h < 38) && (id < Ns);
    for (int c0 = 0; c0 < Cmid; c0 += 8) {
      union { u32x4 u; bf16 b[8]; } t;
      if (valid) t.u = *(const u32x4*)(F + (size_t)id * Cmid + c0);
      else {
#pragma unroll
        for (int i = 0; i < 8; ++i) t.b[i] = (bf16)0.0f;
      }
#pragma unroll
      for (int i = 0; i < 8; ++i) Bt[(c0 + i) * 64 + h] = t.b[i];
    }
  }
  __syncthreads();

  const int  l16 = lane & 15, half = lane >> 4;
  const bool kvalid = l16 < 15;
  float kx = 0.f, ky = 0.f, kz = 0.f;
  if (kvalid) { kx = kpts[l16*3]; ky = kpts[l16*3+1]; kz = kpts[l16*3+2]; }
  const float inv_sig = 1.0f / sigma;

  // build A fragments (influence^T, 16x64 padded) directly in VGPR layout
  bf16x16 afr[2];
#pragma unroll
  for (int chunk = 0; chunk < 2; ++chunk) {
    union { bf16 b[16]; bf16x16 v; } t;
    int kb = chunk * 32 + half * 8;
#pragma unroll
    for (int i = 0; i < 16; ++i) {
      int h = kb + ((i < 8) ? i : i + 8);     // elems 8..15 map to K kb+16..23
      float infl = 0.0f;
      if (kvalid && h < 38) {
        float dx = relx[h]-kx, dy = rely[h]-ky, dz = relz[h]-kz;
        float d  = sqrtf(dx*dx + dy*dy + dz*dz + 1.0e-12f);
        infl = fmaxf(1.0f - d * inv_sig, 0.0f);
      }
      t.b[i] = (bf16)infl;
    }
    afr[chunk] = t.v;
  }

  const int nj = Cmid >> 4;
  for (int j = 0; j < nj; ++j) {
    f32x8 acc = {};
#pragma unroll
    for (int chunk = 0; chunk < 2; ++chunk) {
      union { u32x4 u[2]; bf16x16 v; } b;
      const bf16* p = &Bt[(j*16 + l16) * 64 + chunk*32 + half*8];
      b.u[0] = *(const u32x4*)(p);
      b.u[1] = *(const u32x4*)(p + 16);
      acc = __builtin_amdgcn_wmma_f32_16x16x32_bf16(false, afr[chunk], false, b.v,
                                                    (short)0, acc, false, false);
    }
#pragma unroll
    for (int r = 0; r < 8; ++r) {
      int kk = half * 8 + r;                  // kernel-point row from D layout
      if (kk < 15)
        Agg[(size_t)m * (15 * Cmid) + (size_t)kk * Cmid + j*16 + l16] = (bf16)acc[r];
    }
  }
}

// ---------------- vote tail: off = clip(h @ w2 + b2), shifted = points + off --
__global__ void vote_tail_kernel(const bf16* __restrict__ h, const float* __restrict__ w2,
                                 const float* __restrict__ b2, const float* __restrict__ points,
                                 float* __restrict__ out /* d_out: sp||sa == shifted */)
{
  int i = blockIdx.x * blockDim.x + threadIdx.x;
  if (i >= 65536) return;
  float a0 = b2[0], a1 = b2[1], a2 = b2[2];
  const bf16* hr = h + (size_t)i * 256;
  for (int c = 0; c < 256; ++c) {
    float hv = (float)hr[c];
    a0 += hv * w2[c*3+0]; a1 += hv * w2[c*3+1]; a2 += hv * w2[c*3+2];
  }
  a0 = fminf(fmaxf(a0, -3.f), 3.f);
  a1 = fminf(fmaxf(a1, -3.f), 3.f);
  a2 = fminf(fmaxf(a2, -3.f), 3.f);
  out[(size_t)i*3+0] = points[(size_t)i*3+0] + a0;
  out[(size_t)i*3+1] = points[(size_t)i*3+1] + a1;
  out[(size_t)i*3+2] = points[(size_t)i*3+2] + a2;
}

// ---------------- KNN center averaging ----------------
__global__ void centers_kernel(const float* __restrict__ shifted,
                               const int* __restrict__ pos_knn,
                               const int* __restrict__ anc_knn,
                               float* __restrict__ cen /* pos_c||anc_c (16384x3) */)
{
  int i = blockIdx.x * blockDim.x + threadIdx.x;
  if (i >= 16384) return;
  const int* knn; const float* base; float* o;
  if (i < 8192) { knn = pos_knn + (size_t)i*38; base = shifted;          o = cen + (size_t)i*3; }
  else { int j = i - 8192;
         knn = anc_knn + (size_t)j*38; base = shifted + 98304; o = cen + 24576 + (size_t)j*3; }
  float sx = 0.f, sy = 0.f, sz = 0.f, cnt = 0.f;
  for (int hh = 0; hh < 38; ++hh) {
    int id = knn[hh];
    if (id < 32768) { sx += base[(size_t)id*3]; sy += base[(size_t)id*3+1];
                      sz += base[(size_t)id*3+2]; cnt += 1.f; }
  }
  float inv = 1.f / cnt;
  o[0] = sx*inv; o[1] = sy*inv; o[2] = sz*inv;
}

// ---------------- max-pool shortcut (e1, strided) ----------------
__global__ void maxpool_kernel(const float* __restrict__ feats, const int* __restrict__ sub,
                               float* __restrict__ out)
{
  int m = blockIdx.x; int c = threadIdx.x;   // 16384 x 256
  float best = -1.0e9f;
  for (int hh = 0; hh < 38; ++hh) {
    int id = sub[(size_t)m*38 + hh];
    float v = (id < 65536) ? feats[(size_t)id*256 + c] : -1.0e9f;
    best = fmaxf(best, v);
  }
  out[(size_t)m*256 + c] = best;
}

// ---------------- GroupNorm (global stats over all rows) ----------------
__global__ void gn_stats_kernel(const float* __restrict__ X, int C,
                                float* __restrict__ sums /* [2*C] */)
{
  int c = threadIdx.x;                       // blockDim.x == C
  size_t r0 = (size_t)blockIdx.x * 128;
  const float* p = X + r0 * C + c;
  float s = 0.f, sq = 0.f;
  for (int r = 0; r < 128; ++r) { float v = p[(size_t)r * C]; s += v; sq += v * v; }
  atomicAdd(&sums[c], s);
  atomicAdd(&sums[C + c], sq);
}

__global__ void gn_finalize_kernel(const float* __restrict__ sums, int C, int R,
                                   float* __restrict__ mean, float* __restrict__ rstd)
{
  int g = threadIdx.x;                       // 32 groups
  if (g >= 32) return;
  int cpg = C / 32;
  float s = 0.f, sq = 0.f;
  for (int i = 0; i < cpg; ++i) { s += sums[g*cpg + i]; sq += sums[C + g*cpg + i]; }
  float inv = 1.0f / ((float)R * (float)cpg);
  float m = s * inv, v = sq * inv - m * m;
  mean[g] = m;
  rstd[g] = rsqrtf(v + 1e-5f);
}

__global__ void gn_apply_kernel(const float* __restrict__ X,
                                const float* __restrict__ mean, const float* __restrict__ rstd,
                                const float* __restrict__ gamma, const float* __restrict__ beta,
                                size_t total, int C, int cpg, int act,
                                const float* __restrict__ add,
                                float* __restrict__ outF, bf16* __restrict__ outB)
{
  size_t i = blockIdx.x * (size_t)blockDim.x + threadIdx.x;
  if (i >= total) return;
  int c = (int)(i % (size_t)C);
  int g = c / cpg;
  float v = (X[i] - mean[g]) * rstd[g] * gamma[c] + beta[c];
  if (add) v += add[i];                      // residual before activation
  if (act) v = (v >= 0.0f) ? v : LEAKY_SLOPE * v;
  if (outF) outF[i] = v;
  if (outB) outB[i] = (bf16)v;
}

// ---------------------------------------------------------------------------
extern "C" void kernel_launch(void* const* d_in, const int* in_sizes, int n_in,
                              void* d_out, int out_size, void* d_ws, size_t ws_size,
                              hipStream_t stream)
{
  (void)in_sizes; (void)n_in; (void)out_size; (void)ws_size;
  auto F = [&](int i) { return (const float*)d_in[i]; };
  auto I = [&](int i) { return (const int*)d_in[i]; };

  // setup_inputs() dict order (params nested dict in insertion order)
  const float* feats   = F(0);
  const float* points  = F(1);
  const int* pos_knn   = I(2);
  const int* anc_knn   = I(3);
  const int* subsamp   = I(4);
  const int* neighbors = I(5);
  const float* v_w1 = F(6);  const float* v_b1 = F(7);
  const float* v_w2 = F(8);  const float* v_b2 = F(9);
  const float* e1_u1_w = F(10); const float* e1_u1_b = F(11); const float* e1_u1_g = F(12); const float* e1_u1_be = F(13);
  const float* e1_kp_pts = F(14); const float* e1_kp_w = F(15);
  const float* e1_kn_g = F(16); const float* e1_kn_be = F(17);
  const float* e1_u2_w = F(18); const float* e1_u2_b = F(19); const float* e1_u2_g = F(20); const float* e1_u2_be = F(21);
  const float* e2_u1_w = F(22); const float* e2_u1_b = F(23); const float* e2_u1_g = F(24); const float* e2_u1_be = F(25);
  const float* e2_kp_pts = F(26); const float* e2_kp_w = F(27);
  const float* e2_kn_g = F(28); const float* e2_kn_be = F(29);
  const float* e2_u2_w = F(30); const float* e2_u2_b = F(31); const float* e2_u2_g = F(32); const float* e2_u2_be = F(33);
  const float* e2_sc_w = F(34); const float* e2_sc_b = F(35); const float* e2_sc_g = F(36); const float* e2_sc_be = F(37);
  const float* e3_u1_w = F(38); const float* e3_u1_b = F(39); const float* e3_u1_g = F(40); const float* e3_u1_be = F(41);
  const float* e3_kp_pts = F(42); const float* e3_kp_w = F(43);
  const float* e3_kn_g = F(44); const float* e3_kn_be = F(45);
  const float* e3_u2_w = F(46); const float* e3_u2_b = F(47); const float* e3_u2_g = F(48); const float* e3_u2_be = F(49);

  float* OUT  = (float*)d_out;
  float* CEN  = OUT + 196608;        // pos_c || anc_c -> centers (16384 x 3)
  float* FOUT = OUT + 245760;        // final features (16384 x 512)

  // ---- workspace bump allocator (phase-aliased regions, ~230 MB total) ----
  char* wsb = (char*)d_ws; size_t off = 0;
  auto take = [&](size_t bytes) -> void* {
    void* p = wsb + off; off += (bytes + 255) & ~(size_t)255; return p;
  };
  bf16* wt_vw1  = (bf16*)take((size_t)256*256*2);
  bf16* wt_e1u1 = (bf16*)take((size_t)64*256*2);
  bf16* wt_e1kp = (bf16*)take((size_t)64*960*2);
  bf16* wt_e1u2 = (bf16*)take((size_t)256*64*2);
  bf16* wt_e2u1 = (bf16*)take((size_t)128*256*2);
  bf16* wt_e2kp = (bf16*)take((size_t)128*1920*2);
  bf16* wt_e2u2 = (bf16*)take((size_t)512*128*2);
  bf16* wt_e2sc = (bf16*)take((size_t)512*256*2);
  bf16* wt_e3u1 = (bf16*)take((size_t)128*512*2);
  bf16* wt_e3kp = (bf16*)take((size_t)128*1920*2);
  bf16* wt_e3u2 = (bf16*)take((size_t)512*128*2);
  float* sums  = (float*)take(2*512*sizeof(float));
  float* meanb = (float*)take(32*sizeof(float));
  float* rstdb = (float*)take(32*sizeof(float));

  const size_t MB = 1024ull*1024ull;
  char* R1  = (char*)take(32*MB);                 // featsb -> later t2 (f32 scratch)
  char* R2  = (char*)take(32*MB);                 // hb -> (y1,xnb) -> t3
  bf16* agg = (bf16*)take((size_t)16384*1920*2);  // 60 MB, reused per block
  float* t0  = (float*)take(8*MB);
  bf16*  t0b = (bf16*)take(4*MB);
  float* t1  = (float*)take(16*MB);
  bf16*  t1b = (bf16*)take(4*MB);
  float* f1  = (float*)take(16*MB);
  bf16*  f1b = (bf16*)take(8*MB);
  float* f2  = (float*)take(32*MB);
  bf16*  f2b = (bf16*)take(16*MB);

  bf16*  featsb = (bf16*)R1;          // dead after e1 u1 GEMM
  float* t2     = (float*)R1;         // alive only after that
  bf16*  hb     = (bf16*)R2;          // dead after vote_tail
  float* y1     = (float*)R2;
  bf16*  xnb    = (bf16*)(R2 + 16*MB);
  float* t3     = (float*)R2;         // alive only in e2 (after xnb is dead)

  auto pack = [&](const float* W, bf16* Wt, int K, int N) {
    int total = K*N;
    pack_wt_kernel<<<(total+255)/256, 256, 0, stream>>>(W, Wt, K, N);
  };
  auto gemm = [&](const bf16* A, const bf16* Wt, const float* bias,
                  float* Cf, bf16* Cb, int M, int N, int K, int act) {
    dim3 g(M/128, N/64);
    gemm_bf16_wmma<<<g, 128, 0, stream>>>(A, Wt, bias, Cf, Cb, N, K, act);
  };
  auto gn = [&](const float* X, int R, int C, const float* gamma, const float* beta,
                int act, const float* add, float* oF, bf16* oB) {
    hipMemsetAsync(sums, 0, (size_t)2*C*sizeof(float), stream);
    gn_stats_kernel<<<R/128, C, 0, stream>>>(X, C, sums);
    gn_finalize_kernel<<<1, 32, 0, stream>>>(sums, C, R, meanb, rstdb);
    size_t total = (size_t)R * C;
    gn_apply_kernel<<<(unsigned)((total + 255)/256), 256, 0, stream>>>(
        X, meanb, rstdb, gamma, beta, total, C, C/32, act, add, oF, oB);
  };

  // ---- weight packing + feats -> bf16 ----
  pack(v_w1,    wt_vw1,  256,  256);
  pack(e1_u1_w, wt_e1u1, 256,  64);
  pack(e1_kp_w, wt_e1kp, 960,  64);
  pack(e1_u2_w, wt_e1u2, 64,   256);
  pack(e2_u1_w, wt_e2u1, 256,  128);
  pack(e2_kp_w, wt_e2kp, 1920, 128);
  pack(e2_u2_w, wt_e2u2, 128,  512);
  pack(e2_sc_w, wt_e2sc, 256,  512);
  pack(e3_u1_w, wt_e3u1, 512,  128);
  pack(e3_kp_w, wt_e3kp, 1920, 128);
  pack(e3_u2_w, wt_e3u2, 128,  512);
  { size_t n = (size_t)65536*256;
    f32_to_bf16_kernel<<<(unsigned)((n + 255)/256), 256, 0, stream>>>(feats, featsb, n); }

  // ---- vote MLP: h = relu(feats@w1+b1); shifted -> d_out (sp||sa); centers ----
  gemm(featsb, wt_vw1, v_b1, nullptr, hb, 65536, 256, 256, /*relu*/1);
  vote_tail_kernel<<<256, 256, 0, stream>>>(hb, v_w2, v_b2, points, OUT);
  centers_kernel<<<64, 256, 0, stream>>>(OUT, pos_knn, anc_knn, CEN);

  // ---- block e1 (256 -> 64 -> 256, strided max-pool shortcut, sigma=1) ----
  gemm(featsb, wt_e1u1, e1_u1_b, y1, nullptr, 65536, 64, 256, 0);
  gn(y1, 65536, 64, e1_u1_g, e1_u1_be, 1, nullptr, nullptr, xnb);
  kpconv_agg_kernel<<<16384, 32, 0, stream>>>(points, 65536, CEN, xnb, 64,
                                              subsamp, e1_kp_pts, 1.0f, agg);
  gemm(agg, wt_e1kp, nullptr, t0, nullptr, 16384, 64, 960, 0);
  gn(t0, 16384, 64, e1_kn_g, e1_kn_be, 1, nullptr, nullptr, t0b);
  gemm(t0b, wt_e1u2, e1_u2_b, t1, nullptr, 16384, 256, 64, 0);
  maxpool_kernel<<<16384, 256, 0, stream>>>(feats, subsamp, t2);
  gn(t1, 16384, 256, e1_u2_g, e1_u2_be, 1, t2, f1, f1b);

  // ---- block e2 (256 -> 128 -> 512, GN shortcut, sigma=2) ----
  gemm(f1b, wt_e2u1, e2_u1_b, t0, nullptr, 16384, 128, 256, 0);
  gn(t0, 16384, 128, e2_u1_g, e2_u1_be, 1, nullptr, nullptr, t0b);
  kpconv_agg_kernel<<<16384, 32, 0, stream>>>(CEN, 16384, CEN, t0b, 128,
                                              neighbors, e2_kp_pts, 2.0f, agg);
  gemm(agg, wt_e2kp, nullptr, t1, nullptr, 16384, 128, 1920, 0);
  gn(t1, 16384, 128, e2_kn_g, e2_kn_be, 1, nullptr, nullptr, t1b);
  gemm(t1b, wt_e2u2, e2_u2_b, t2, nullptr, 16384, 512, 128, 0);
  gemm(f1b, wt_e2sc, e2_sc_b, t3, nullptr, 16384, 512, 256, 0);
  gn(t3, 16384, 512, e2_sc_g, e2_sc_be, 0, nullptr, t3, nullptr);   // in-place
  gn(t2, 16384, 512, e2_u2_g, e2_u2_be, 1, t3, f2, f2b);

  // ---- block e3 (512 -> 128 -> 512, identity shortcut, sigma=2) ----
  gemm(f2b, wt_e3u1, e3_u1_b, t0, nullptr, 16384, 128, 512, 0);
  gn(t0, 16384, 128, e3_u1_g, e3_u1_be, 1, nullptr, nullptr, t0b);
  kpconv_agg_kernel<<<16384, 32, 0, stream>>>(CEN, 16384, CEN, t0b, 128,
                                              neighbors, e3_kp_pts, 2.0f, agg);
  gemm(agg, wt_e3kp, nullptr, t1, nullptr, 16384, 128, 1920, 0);
  gn(t1, 16384, 128, e3_kn_g, e3_kn_be, 1, nullptr, nullptr, t1b);
  gemm(t1b, wt_e3u2, e3_u2_b, t2, nullptr, 16384, 512, 128, 0);
  gn(t2, 16384, 512, e3_u2_g, e3_u2_be, 1, f2, FOUT, nullptr);
}
